// Head_57921928954112
// MI455X (gfx1250) — matrix operand order repaired and checked by reading, not compiled, not run on hardware
//
#include <hip/hip_runtime.h>
#include <stdint.h>

// Shapes from the reference
#define S_LEN 1024
#define E_DIM 768
#define HD    64
#define NB    32

typedef __attribute__((ext_vector_type(16))) _Float16     v16h;
typedef __attribute__((ext_vector_type(2)))  _Float16     v2h;
typedef __attribute__((ext_vector_type(8)))  float        v8f;
typedef __attribute__((ext_vector_type(4)))  float        v4f;
typedef __attribute__((ext_vector_type(2)))  unsigned int v2u;
typedef __attribute__((ext_vector_type(4)))  unsigned int v4u;
typedef __attribute__((ext_vector_type(8)))  unsigned int v8u;

static __device__ __forceinline__ unsigned short f2h(float f) {
  return __builtin_bit_cast(unsigned short, (_Float16)f);
}
// Pack two f32 -> packed f16 pair in one v_cvt_pk_rtz_f16_f32
static __device__ __forceinline__ unsigned pkrtz(float lo, float hi) {
  return __builtin_bit_cast(unsigned, __builtin_amdgcn_cvt_pkrtz(lo, hi));
}
static __device__ __forceinline__ v8f vzero8() {
  return (v8f){0.f, 0.f, 0.f, 0.f, 0.f, 0.f, 0.f, 0.f};
}
static __device__ __forceinline__ v8f wmma16(v16h a, v16h b, v8f c) {
  // D = A(16x32 f16) * B(32x16 f16) + C(16x16 f32)
  return __builtin_amdgcn_wmma_f32_16x16x32_f16(false, a, false, b, (short)0, c,
                                                false, false);
}
// Build a 16-half fragment from two 16-byte runs (8 halves each)
static __device__ __forceinline__ v16h ld_frag16(const unsigned short* p0,
                                                 const unsigned short* p1) {
  v4u x = *(const v4u*)p0;
  v4u y = *(const v4u*)p1;
  v8u r;
  r[0] = x[0]; r[1] = x[1]; r[2] = x[2]; r[3] = x[3];
  r[4] = y[0]; r[5] = y[1]; r[6] = y[2]; r[7] = y[3];
  return __builtin_bit_cast(v16h, r);
}

// ---------------------------------------------------------------------------
// Stage 0: one-shot weight conversion f32 -> f16 (3 x [64 x 768], 288 KB).
// Removes per-M-tile re-conversion of weights from the stage-1 inner loop.
// ---------------------------------------------------------------------------
__global__ __launch_bounds__(256) void cvt_weights(
    const float* __restrict__ Wq, const float* __restrict__ Wk,
    const float* __restrict__ Wv, unsigned short* __restrict__ w16) {
  const int n = HD * E_DIM;  // 49152 per weight
  const int i = (blockIdx.x * 256 + threadIdx.x) * 4;
  const float* src = (blockIdx.y == 0) ? Wq : (blockIdx.y == 1) ? Wk : Wv;
  unsigned short* dst = w16 + (size_t)blockIdx.y * n;
  v4f f = *(const v4f*)(src + i);
  v2u p;
  p[0] = pkrtz(f[0], f[1]);
  p[1] = pkrtz(f[2], f[3]);
  *(v2u*)(dst + i) = p;
}

// ---------------------------------------------------------------------------
// Stage 1: q/k/v projections (WMMA f16, f32 accum) + RoPE on q,k.
// One wave = one 16-row M tile of x; N = 64 (4 ntiles); K = 768 in steps of 32.
// Outputs: rq,rk row-major [B*S,64] f16 ; v transposed per batch [B][64][S] f16.
// ---------------------------------------------------------------------------
__global__ __launch_bounds__(256) void qkv_rope(
    const float* __restrict__ x, const unsigned short* __restrict__ w16,
    unsigned short* __restrict__ rq, unsigned short* __restrict__ rk,
    unsigned short* __restrict__ vt) {
  const int lane = threadIdx.x & 31;
  const int wvid = threadIdx.x >> 5;
  const int m0 = (blockIdx.x * 8 + wvid) * 16;  // global row tile start
  const int ln = lane & 15;                     // A-row / B-col / D-col index
  const int lg = lane >> 4;                     // lane group (K split / row+8)

  v8f acc[3][4];
#pragma unroll
  for (int w = 0; w < 3; ++w)
#pragma unroll
    for (int nt = 0; nt < 4; ++nt) acc[w][nt] = vzero8();

  const unsigned short* Ws16[3] = {w16, w16 + HD * E_DIM, w16 + 2 * HD * E_DIM};
  const float* xrow = x + (size_t)(m0 + ln) * E_DIM;

  for (int k0 = 0; k0 < E_DIM; k0 += 32) {
    // A fragment: halves 0..7 -> K = k0+lg*8+h ; halves 8..15 -> K = k0+16+lg*8+h
    const float* pa = xrow + k0 + lg * 8;
    v4f f0 = *(const v4f*)(pa);
    v4f f1 = *(const v4f*)(pa + 4);
    v4f f2 = *(const v4f*)(pa + 16);
    v4f f3 = *(const v4f*)(pa + 20);
    v8u au;
    au[0] = pkrtz(f0[0], f0[1]); au[1] = pkrtz(f0[2], f0[3]);
    au[2] = pkrtz(f1[0], f1[1]); au[3] = pkrtz(f1[2], f1[3]);
    au[4] = pkrtz(f2[0], f2[1]); au[5] = pkrtz(f2[2], f2[3]);
    au[6] = pkrtz(f3[0], f3[1]); au[7] = pkrtz(f3[2], f3[3]);
    v16h a = __builtin_bit_cast(v16h, au);
#pragma unroll
    for (int w = 0; w < 3; ++w) {
#pragma unroll
      for (int nt = 0; nt < 4; ++nt) {
        // B fragment: B[k][n] = W[nt*16+n][k]; K = k0+lg*8+h and k0+16+lg*8+h -> 2x16B loads
        const unsigned short* pb =
            Ws16[w] + (size_t)(nt * 16 + ln) * E_DIM + k0 + lg * 8;
        v16h b = ld_frag16(pb, pb + 16);
        acc[w][nt] = wmma16(a, b, acc[w][nt]);
      }
    }
  }

  // Epilogue: RoPE on q,k in the D layout (row = r + 8*lg, col h = nt*16+ln)
  const int batch = m0 >> 10;                  // S == 1024
  const int s0 = (m0 & (S_LEN - 1)) + lg * 8;  // seq position of element r=0
#pragma unroll
  for (int nt = 0; nt < 4; ++nt) {
    const int h = nt * 16 + ln;
    // theta_j = 10000^(-j/16), j = h>>1 ; ln(10000)/16 = 0.575646273...
    const float theta = expf(-(float)(h >> 1) * 0.57564627324851142f);
    const float sg = (h & 1) ? 1.0f : -1.0f;
    float vv[8];
#pragma unroll
    for (int r = 0; r < 8; ++r) {
      float cn, sn;
      sincosf((float)(s0 + r) * theta, &sn, &cn);
      float qv = acc[0][nt][r];
      float kv = acc[1][nt][r];
      float qp = __shfl_xor(qv, 1, 32);  // partner head-dim (h^1) lives in lane^1
      float kp = __shfl_xor(kv, 1, 32);
      float qr = qv * cn + sg * (qp * sn);
      float kr = kv * cn + sg * (kp * sn);
      size_t off = (size_t)(m0 + lg * 8 + r) * HD + h;
      rq[off] = f2h(qr);
      rk[off] = f2h(kr);
      vv[r] = acc[2][nt][r];
    }
    // v transposed: vT[batch][h][s], 8 consecutive s -> one 16B store
    v4u pv;
    pv[0] = pkrtz(vv[0], vv[1]);
    pv[1] = pkrtz(vv[2], vv[3]);
    pv[2] = pkrtz(vv[4], vv[5]);
    pv[3] = pkrtz(vv[6], vv[7]);
    *(v4u*)(vt + ((size_t)(batch * HD + h) * S_LEN + s0)) = pv;
  }
}

// ---------------------------------------------------------------------------
// Stage 2: causal flash attention. One wave = one 16-query tile of one batch.
// Loop over 32-key strips; QK^T via WMMA; online softmax with shfl reductions;
// P re-layout via per-wave LDS bounce; PV via WMMA against transposed V.
// ---------------------------------------------------------------------------
__global__ __launch_bounds__(256) void flash_attn(
    const unsigned short* __restrict__ rq, const unsigned short* __restrict__ rk,
    const unsigned short* __restrict__ vt, float* __restrict__ out) {
  __shared__ __align__(16) unsigned short pl[8][16][32];  // P bounce per wave
  const int lane = threadIdx.x & 31;
  const int wvid = threadIdx.x >> 5;
  const int wid = blockIdx.x * 8 + wvid;  // 2048 waves total
  const int batch = wid >> 6;
  const int q0 = (wid & 63) * 16;
  const int ln = lane & 15;
  const int lg = lane >> 4;

  // Q A-fragments for K-dim (head dim) 0..31 and 32..63
  const unsigned short* qr = rq + (size_t)(batch * S_LEN + q0 + ln) * HD;
  v16h aq0 = ld_frag16(qr + lg * 8, qr + 16 + lg * 8);
  v16h aq1 = ld_frag16(qr + 32 + lg * 8, qr + 48 + lg * 8);

  v8f O[4];
#pragma unroll
  for (int nt = 0; nt < 4; ++nt) O[nt] = vzero8();
  float Mr[8], Lr[8];
#pragma unroll
  for (int r = 0; r < 8; ++r) { Mr[r] = -1e30f; Lr[r] = 0.f; }

  const int qmax = q0 + 15;
  const int nstrips = (qmax >> 5) + 1;
  for (int ks = 0; ks < nstrips; ++ks) {
    const int k0 = ks << 5;
    float st[2][8];
#pragma unroll
    for (int t = 0; t < 2; ++t) {
      const int kb = k0 + t * 16;
      if (kb <= qmax) {  // wave-uniform guard
        const unsigned short* kr = rk + (size_t)(batch * S_LEN + kb + ln) * HD;
        v16h b0 = ld_frag16(kr + lg * 8, kr + 16 + lg * 8);
        v16h b1 = ld_frag16(kr + 32 + lg * 8, kr + 48 + lg * 8);
        v8f s = vzero8();
        s = wmma16(aq0, b0, s);
        s = wmma16(aq1, b1, s);
#pragma unroll
        for (int r = 0; r < 8; ++r) {
          float sv = s[r] * 0.03608439182435161f;          // 768^-0.5
          if (kb + ln > q0 + r + lg * 8) sv = -1e30f;      // causal mask
          st[t][r] = sv;
        }
      } else {
#pragma unroll
        for (int r = 0; r < 8; ++r) st[t][r] = -1e30f;
      }
    }
    // Online softmax: rows live in 16-lane groups -> xor 1,2,4,8 reductions
    float al[8], p0a[8], p1a[8];
#pragma unroll
    for (int r = 0; r < 8; ++r) {
      float mx = fmaxf(st[0][r], st[1][r]);
      mx = fmaxf(mx, __shfl_xor(mx, 1, 32));
      mx = fmaxf(mx, __shfl_xor(mx, 2, 32));
      mx = fmaxf(mx, __shfl_xor(mx, 4, 32));
      mx = fmaxf(mx, __shfl_xor(mx, 8, 32));
      float mnew = fmaxf(Mr[r], mx);
      float a_ = __expf(Mr[r] - mnew);
      float p0 = __expf(st[0][r] - mnew);
      float p1 = __expf(st[1][r] - mnew);
      float rs = p0 + p1;
      rs += __shfl_xor(rs, 1, 32);
      rs += __shfl_xor(rs, 2, 32);
      rs += __shfl_xor(rs, 4, 32);
      rs += __shfl_xor(rs, 8, 32);
      Lr[r] = Lr[r] * a_ + rs;
      Mr[r] = mnew; al[r] = a_; p0a[r] = p0; p1a[r] = p1;
    }
#pragma unroll
    for (int nt = 0; nt < 4; ++nt)
#pragma unroll
      for (int r = 0; r < 8; ++r) O[nt][r] = O[nt][r] * al[r];
    // P (D layout) -> LDS [row][keycol] f16 ; DS ops are in-order per wave
#pragma unroll
    for (int r = 0; r < 8; ++r) {
      pl[wvid][lg * 8 + r][ln]      = f2h(p0a[r]);
      pl[wvid][lg * 8 + r][16 + ln] = f2h(p1a[r]);
    }
    // Re-read as A fragment (16 queries x 32 keys)
    v16h ap = ld_frag16(&pl[wvid][ln][lg * 8], &pl[wvid][ln][16 + lg * 8]);
    // PV: B[k][n] = v[k0+k][nt*16+n] = vT[batch][nt*16+n][k0+k] (contiguous)
#pragma unroll
    for (int nt = 0; nt < 4; ++nt) {
      const unsigned short* vr =
          vt + (size_t)(batch * HD + nt * 16 + ln) * S_LEN + k0 + lg * 8;
      v16h bv = ld_frag16(vr, vr + 16);
      O[nt] = wmma16(ap, bv, O[nt]);
    }
  }
  // Normalize and store (f32)
#pragma unroll
  for (int nt = 0; nt < 4; ++nt)
#pragma unroll
    for (int r = 0; r < 8; ++r)
      out[(size_t)(batch * S_LEN + q0 + lg * 8 + r) * HD + nt * 16 + ln] =
          O[nt][r] / Lr[r];
}

extern "C" void kernel_launch(void* const* d_in, const int* in_sizes, int n_in,
                              void* d_out, int out_size, void* d_ws,
                              size_t ws_size, hipStream_t stream) {
  const float* x  = (const float*)d_in[0];
  const float* Wq = (const float*)d_in[1];
  const float* Wk = (const float*)d_in[2];
  const float* Wv = (const float*)d_in[3];
  float* outp = (float*)d_out;

  const size_t qkv_elems = (size_t)NB * S_LEN * HD;  // 2M f16 each
  unsigned short* rq  = (unsigned short*)d_ws;
  unsigned short* rk  = rq + qkv_elems;
  unsigned short* vt  = rk + qkv_elems;
  unsigned short* w16 = vt + qkv_elems;  // 3 x 64 x 768 f16 (288 KB)

  // Stage 0: weights f32 -> f16 once (48 blocks x 3 weights, 4 elems/thread)
  cvt_weights<<<dim3(48, 3), 256, 0, stream>>>(Wq, Wk, Wv, w16);
  // Stage 1: 2048 M-tiles (32768 rows / 16), 8 waves per 256-thread block
  qkv_rope<<<256, 256, 0, stream>>>(x, w16, rq, rk, vt);
  // Stage 2: 2048 query tiles (32 batches * 64 tiles), 8 waves per block
  flash_attn<<<256, 256, 0, stream>>>(rq, rk, vt, outp);
}
